// MambaBlock_76699525971984
// MI455X (gfx1250) — compile-verified, matrix operand order checked by reading, not run on hardware
//
#include <hip/hip_runtime.h>

// ---------------- problem dims ----------------
#define B_SZ 2
#define L_SZ 2048
#define DMODEL 1024
#define DINNER 2048
#define DSTATE 16
#define DTRANK 64
#define DCONV 4
#define NTOK (B_SZ * L_SZ)            // 4096 tokens
#define XDBL_W (DTRANK + 2 * DSTATE)  // 96
#define NCHUNK 16
#define CHUNK (L_SZ / NCHUNK)         // 128

typedef __bf16 bf16;
typedef __attribute__((ext_vector_type(16))) __bf16 v16bf;
typedef __attribute__((ext_vector_type(8)))  __bf16 v8bf;
typedef __attribute__((ext_vector_type(8)))  float  v8f;

__device__ __forceinline__ float sigmoidf_(float x) { return 1.f / (1.f + __expf(-x)); }
__device__ __forceinline__ float siluf_(float x)    { return x * sigmoidf_(x); }
__device__ __forceinline__ float softplusf_(float x){ return x > 20.f ? x : __logf(1.f + __expf(x)); }

// ---------------- f32 -> bf16 conversion (operand staging for WMMA GEMMs) ----------------
__global__ __launch_bounds__(256)
void cvt_f32_bf16(const float* __restrict__ s, bf16* __restrict__ d, int n)
{
    const int i = (blockIdx.x * 256 + threadIdx.x) * 4;
    if (i + 3 < n) {
        float4 f = *(const float4*)(s + i);
        d[i]     = (bf16)f.x;
        d[i + 1] = (bf16)f.y;
        d[i + 2] = (bf16)f.z;
        d[i + 3] = (bf16)f.w;
    }
}

// ---------------- bf16 WMMA GEMM (NT): C[M,N] = A[M,K] * W[N,K]^T ----------------
// Block tile 128x128, K-step 32, 256 threads = 8 waves (wave32), wave tile 64x32.
// Double-buffered LDS fed by GLOBAL_LOAD_ASYNC_TO_LDS_B128 (ASYNCcnt pipeline).
// EPI: 0 = plain f32 store, 1 = softplus(acc + bias[n])
#define BM 128
#define BN 128
#define BK 32
#define LDSS 40   // bf16 row stride: 80B, 16B-aligned

__device__ __forceinline__ unsigned lds_addr32(const void* p)
{
    // generic (flat) LDS pointer: low 32 bits are the LDS byte address
    return (unsigned)(unsigned long long)p;
}

// Issue 2 async b128 copies: 16 contiguous bf16 (one row-half) global -> LDS.
__device__ __forceinline__ void async_row32(const bf16* gbase, unsigned gByteOff, unsigned ldsByte)
{
    asm volatile(
        "global_load_async_to_lds_b128 %0, %1, %4\n\t"
        "global_load_async_to_lds_b128 %2, %3, %4"
        :
        : "v"(ldsByte), "v"(gByteOff),
          "v"(ldsByte + 16u), "v"(gByteOff + 16u),
          "s"(gbase)
        : "memory");
}

__device__ __forceinline__ void wait_async_le4()
{
    asm volatile("s_wait_asynccnt 0x4" ::: "memory");
}
__device__ __forceinline__ void wait_async_0()
{
    asm volatile("s_wait_asynccnt 0x0" ::: "memory");
}

template <int EPI>
__global__ __launch_bounds__(256)
void gemm_nt_bf16(const bf16* __restrict__ A, int lda,
                  const bf16* __restrict__ W, int ldb,
                  float* __restrict__ C, int ldc,
                  const float* __restrict__ bias,
                  int M, int N, int K)
{
    __shared__ __align__(16) bf16 sA[2][BM * LDSS];
    __shared__ __align__(16) bf16 sB[2][BN * LDSS];

    const int tid  = threadIdx.x;
    const int lane = tid & 31;
    const int wave = tid >> 5;
    const int wm   = wave >> 2;            // 0..1  -> 64-row slab
    const int wn   = wave & 3;             // 0..3  -> 32-col slab
    const int mTile = blockIdx.y * BM;
    const int nTile = blockIdx.x * BN;

    // tile loader mapping: 2 threads per row, 16 bf16 (= 2 x b128) each
    const int tRow = tid >> 1;             // 0..127
    const int tCol = (tid & 1) << 4;       // 0 or 16 (bf16 elements)

    // clamp OOB rows to a valid address (results masked at store)
    int ga = mTile + tRow; if (ga > M - 1) ga = M - 1;
    int gb = nTile + tRow; if (gb > N - 1) gb = N - 1;
    const unsigned aOffBase = ((unsigned)ga * (unsigned)lda + (unsigned)tCol) * 2u;
    const unsigned bOffBase = ((unsigned)gb * (unsigned)ldb + (unsigned)tCol) * 2u;
    const unsigned ldsAByte0 = lds_addr32(&sA[0][tRow * LDSS + tCol]);
    const unsigned ldsBByte0 = lds_addr32(&sB[0][tRow * LDSS + tCol]);
    const unsigned ldsAByte1 = lds_addr32(&sA[1][tRow * LDSS + tCol]);
    const unsigned ldsBByte1 = lds_addr32(&sB[1][tRow * LDSS + tCol]);

    // WMMA fragment lane mapping (16-bit A/B layout, 16x16x32)
    const int fr      = lane & 15;
    const int rowHalf = lane >> 4;
    const int khalf   = rowHalf << 3;

    v8f acc[4][2];
#pragma unroll
    for (int mf = 0; mf < 4; ++mf)
#pragma unroll
        for (int nf = 0; nf < 2; ++nf)
            acc[mf][nf] = v8f{0.f, 0.f, 0.f, 0.f, 0.f, 0.f, 0.f, 0.f};

    const int nIter = K / BK;
    // prologue: stage tile 0 into buffer 0 (4 async instructions / thread)
    async_row32(A, aOffBase, ldsAByte0);
    async_row32(W, bOffBase, ldsBByte0);

    for (int it = 0; it < nIter; ++it) {
        const int cur = it & 1;
        // stage next tile into the other buffer (safe: all waves passed the
        // end-of-iteration barrier of it-1, so nobody still reads it)
        if (it + 1 < nIter) {
            const unsigned koff = (unsigned)(it + 1) * (BK * 2u);
            async_row32(A, aOffBase + koff, cur ? ldsAByte0 : ldsAByte1);
            async_row32(W, bOffBase + koff, cur ? ldsBByte0 : ldsBByte1);
            wait_async_le4();   // current tile complete; next 4 still in flight
        } else {
            wait_async_0();
        }
        __syncthreads();

        v16bf aF[4], bF[2];
#pragma unroll
        for (int mf = 0; mf < 4; ++mf) {
            const bf16* p = &sA[cur][(wm * 64 + mf * 16 + fr) * LDSS + khalf];
            v8bf lo = *(const v8bf*)p;
            v8bf hi = *(const v8bf*)(p + 16);
            aF[mf] = __builtin_shufflevector(lo, hi, 0,1,2,3,4,5,6,7,8,9,10,11,12,13,14,15);
        }
#pragma unroll
        for (int nf = 0; nf < 2; ++nf) {
            const bf16* p = &sB[cur][(wn * 32 + nf * 16 + fr) * LDSS + khalf];
            v8bf lo = *(const v8bf*)p;
            v8bf hi = *(const v8bf*)(p + 16);
            bF[nf] = __builtin_shufflevector(lo, hi, 0,1,2,3,4,5,6,7,8,9,10,11,12,13,14,15);
        }
#pragma unroll
        for (int mf = 0; mf < 4; ++mf)
#pragma unroll
            for (int nf = 0; nf < 2; ++nf)
                acc[mf][nf] = __builtin_amdgcn_wmma_f32_16x16x32_bf16(
                    false, aF[mf], false, bF[nf], (short)0, acc[mf][nf], false, false);
        __syncthreads();
    }

    // ---- epilogue / store (C/D layout: lane=col%16, vgpr e -> row e+8*rowHalf) ----
#pragma unroll
    for (int mf = 0; mf < 4; ++mf)
#pragma unroll
        for (int nf = 0; nf < 2; ++nf) {
            const int col = nTile + wn * 32 + nf * 16 + fr;
            if (col < N) {
                float bv = (EPI == 1) ? bias[col] : 0.f;
#pragma unroll
                for (int e = 0; e < 8; ++e) {
                    const int row = mTile + wm * 64 + mf * 16 + rowHalf * 8 + e;
                    if (row < M) {
                        float val = acc[mf][nf][e];
                        if (EPI == 1) val = softplusf_(val + bv);
                        C[(size_t)row * ldc + col] = val;
                    }
                }
            }
        }
}

// ---------------- causal depthwise conv (k=4) + SiLU -> u (f32 + bf16) ----------------
__global__ __launch_bounds__(256)
void conv_silu_kernel(const float* __restrict__ xz, const float* __restrict__ cw,
                      const float* __restrict__ cb, float* __restrict__ u,
                      bf16* __restrict__ u_bf)
{
    const int idx = blockIdx.x * 256 + threadIdx.x;        // over NTOK*DINNER
    const int d   = idx & (DINNER - 1);
    const int row = idx >> 11;                             // b*L + t
    const int t   = row & (L_SZ - 1);
    float acc = cb[d];
#pragma unroll
    for (int j = 0; j < DCONV; ++j) {
        const int tt = t - (DCONV - 1) + j;
        if (tt >= 0)
            acc += cw[d * DCONV + j] * xz[(size_t)(row - (DCONV - 1) + j) * (2 * DINNER) + d];
    }
    const float v = siluf_(acc);
    u[idx] = v;
    u_bf[idx] = (bf16)v;
}

// ---------------- selective scan, phase A: per-(channel,chunk) local scan ----------------
__global__ __launch_bounds__(256)
void scan_chunk_kernel(const float* __restrict__ delta, const float* __restrict__ u,
                       const float* __restrict__ xdbl, const float* __restrict__ A_log,
                       float* __restrict__ chP, float* __restrict__ chH)
{
    const int d = blockIdx.x * 256 + threadIdx.x;  // 0..DINNER-1
    const int c = blockIdx.y;                      // chunk
    const int b = blockIdx.z;
    float An[DSTATE];
#pragma unroll
    for (int n = 0; n < DSTATE; ++n) An[n] = -__expf(A_log[d * DSTATE + n]);
    float h[DSTATE], P[DSTATE];
#pragma unroll
    for (int n = 0; n < DSTATE; ++n) { h[n] = 0.f; P[n] = 1.f; }

    for (int i = 0; i < CHUNK; ++i) {
        const int row = b * L_SZ + c * CHUNK + i;
        const float dt = delta[(size_t)row * DINNER + d];
        const float du = dt * u[(size_t)row * DINNER + d];
        const float* Bv = xdbl + (size_t)row * XDBL_W + DTRANK;
#pragma unroll
        for (int n = 0; n < DSTATE; ++n) {
            const float dA = __expf(dt * An[n]);
            h[n] = dA * h[n] + du * Bv[n];
            P[n] *= dA;
        }
    }
    const size_t base = (((size_t)(b * DINNER + d)) * NCHUNK + c) * DSTATE;
#pragma unroll
    for (int n = 0; n < DSTATE; ++n) { chP[base + n] = P[n]; chH[base + n] = h[n]; }
}

// ---------------- phase B: sequential combine over 16 chunks per channel ----------------
__global__ __launch_bounds__(256)
void scan_combine_kernel(const float* __restrict__ chP, const float* __restrict__ chH,
                         float* __restrict__ hstart)
{
    const int idx = blockIdx.x * 256 + threadIdx.x;  // b*DINNER + d
    float hs[DSTATE];
#pragma unroll
    for (int n = 0; n < DSTATE; ++n) hs[n] = 0.f;
    for (int c = 0; c < NCHUNK; ++c) {
        const size_t base = ((size_t)idx * NCHUNK + c) * DSTATE;
#pragma unroll
        for (int n = 0; n < DSTATE; ++n) {
            hstart[base + n] = hs[n];
            hs[n] = chP[base + n] * hs[n] + chH[base + n];
        }
    }
}

// ---- phase C: re-scan with correct init, fuse y = h.C + u*D, gate silu(res), emit bf16 ----
__global__ __launch_bounds__(256)
void scan_output_kernel(const float* __restrict__ delta, const float* __restrict__ u,
                        const float* __restrict__ xdbl, const float* __restrict__ A_log,
                        const float* __restrict__ hstart, const float* __restrict__ xz,
                        const float* __restrict__ Dvec, bf16* __restrict__ yg_bf)
{
    const int d = blockIdx.x * 256 + threadIdx.x;
    const int c = blockIdx.y;
    const int b = blockIdx.z;
    float An[DSTATE];
#pragma unroll
    for (int n = 0; n < DSTATE; ++n) An[n] = -__expf(A_log[d * DSTATE + n]);
    float h[DSTATE];
    {
        const size_t base = (((size_t)(b * DINNER + d)) * NCHUNK + c) * DSTATE;
#pragma unroll
        for (int n = 0; n < DSTATE; ++n) h[n] = hstart[base + n];
    }
    const float Dd = Dvec[d];
    for (int i = 0; i < CHUNK; ++i) {
        const int row = b * L_SZ + c * CHUNK + i;
        const float dt = delta[(size_t)row * DINNER + d];
        const float uu = u[(size_t)row * DINNER + d];
        const float du = dt * uu;
        const float* Bv = xdbl + (size_t)row * XDBL_W + DTRANK;
        const float* Cv = xdbl + (size_t)row * XDBL_W + DTRANK + DSTATE;
        float y = 0.f;
#pragma unroll
        for (int n = 0; n < DSTATE; ++n) {
            const float dA = __expf(dt * An[n]);
            h[n] = dA * h[n] + du * Bv[n];
            y += h[n] * Cv[n];
        }
        const float res = xz[(size_t)row * (2 * DINNER) + DINNER + d];
        yg_bf[(size_t)row * DINNER + d] = (bf16)((y + uu * Dd) * siluf_(res));
    }
}

// ---------------- launcher ----------------
extern "C" void kernel_launch(void* const* d_in, const int* in_sizes, int n_in,
                              void* d_out, int out_size, void* d_ws, size_t ws_size,
                              hipStream_t stream) {
    const float* x         = (const float*)d_in[0];
    const float* in_proj_w = (const float*)d_in[1];
    const float* conv_w    = (const float*)d_in[2];
    const float* conv_b    = (const float*)d_in[3];
    const float* x_proj_w  = (const float*)d_in[4];
    const float* dt_proj_w = (const float*)d_in[5];
    const float* dt_proj_b = (const float*)d_in[6];
    const float* A_log     = (const float*)d_in[7];
    const float* Dvec      = (const float*)d_in[8];
    const float* out_proj_w= (const float*)d_in[9];
    float* out = (float*)d_out;

    // ---- f32 workspace ----
    float* ws = (float*)d_ws;
    size_t off = 0;
    float* xz    = ws + off; off += (size_t)NTOK * 2 * DINNER;             // 64 MB
    float* u     = ws + off; off += (size_t)NTOK * DINNER;                 // 32 MB
    float* xdbl  = ws + off; off += (size_t)NTOK * XDBL_W;                 // 1.5 MB
    float* delta = ws + off; off += (size_t)NTOK * DINNER;                 // 32 MB
    float* chP   = ws + off; off += (size_t)B_SZ * DINNER * NCHUNK * DSTATE;
    float* chH   = ws + off; off += (size_t)B_SZ * DINNER * NCHUNK * DSTATE;
    float* hst   = ws + off; off += (size_t)B_SZ * DINNER * NCHUNK * DSTATE;
    // ---- bf16 workspace (16B-aligned: all prior sizes are multiples of 8 floats) ----
    bf16* bws = (bf16*)(ws + off);
    size_t boff = 0;
    bf16* x_bf    = bws + boff; boff += (size_t)NTOK * DMODEL;
    bf16* wi_bf   = bws + boff; boff += (size_t)(2 * DINNER) * DMODEL;
    bf16* wx_bf   = bws + boff; boff += (size_t)XDBL_W * DINNER;
    bf16* wd_bf   = bws + boff; boff += (size_t)DINNER * DTRANK;
    bf16* wo_bf   = bws + boff; boff += (size_t)DMODEL * DINNER;
    bf16* u_bf    = bws + boff; boff += (size_t)NTOK * DINNER;
    bf16* xdbl_bf = bws + boff; boff += (size_t)NTOK * XDBL_W;
    bf16* yg_bf   = bws + boff; boff += (size_t)NTOK * DINNER;

    dim3 blk(256);

    // 0) stage bf16 operands (weights + x); each thread converts 4 elements
    {
        int n;
        n = NTOK * DMODEL;            cvt_f32_bf16<<<n / 1024, blk, 0, stream>>>(x, x_bf, n);
        n = 2 * DINNER * DMODEL;      cvt_f32_bf16<<<n / 1024, blk, 0, stream>>>(in_proj_w, wi_bf, n);
        n = XDBL_W * DINNER;          cvt_f32_bf16<<<n / 1024, blk, 0, stream>>>(x_proj_w, wx_bf, n);
        n = DINNER * DTRANK;          cvt_f32_bf16<<<n / 1024, blk, 0, stream>>>(dt_proj_w, wd_bf, n);
        n = DMODEL * DINNER;          cvt_f32_bf16<<<n / 1024, blk, 0, stream>>>(out_proj_w, wo_bf, n);
    }

    // 1) in_proj: xz[4096, 4096] = x * in_proj_w^T
    gemm_nt_bf16<0><<<dim3((2 * DINNER) / BN, NTOK / BM), blk, 0, stream>>>(
        x_bf, DMODEL, wi_bf, DMODEL, xz, 2 * DINNER, nullptr, NTOK, 2 * DINNER, DMODEL);

    // 2) causal depthwise conv + SiLU -> u (f32 for scan, bf16 for GEMM)
    conv_silu_kernel<<<(NTOK * DINNER) / 256, blk, 0, stream>>>(xz, conv_w, conv_b, u, u_bf);

    // 3) x_proj: xdbl[4096, 96] = u * x_proj_w^T ; then bf16 copy of xdbl for dt_proj
    gemm_nt_bf16<0><<<dim3((XDBL_W + BN - 1) / BN, NTOK / BM), blk, 0, stream>>>(
        u_bf, DINNER, wx_bf, DINNER, xdbl, XDBL_W, nullptr, NTOK, XDBL_W, DINNER);
    {
        int n = NTOK * XDBL_W;
        cvt_f32_bf16<<<n / 1024, blk, 0, stream>>>(xdbl, xdbl_bf, n);
    }

    // 4) dt_proj + bias + softplus: delta[4096,2048] (A = xdbl_bf[:, :64], lda = 96)
    gemm_nt_bf16<1><<<dim3(DINNER / BN, NTOK / BM), blk, 0, stream>>>(
        xdbl_bf, XDBL_W, wd_bf, DTRANK, delta, DINNER, dt_proj_b, NTOK, DINNER, DTRANK);

    // 5-7) chunked selective scan (fused discretization, skip, gate)
    scan_chunk_kernel<<<dim3(DINNER / 256, NCHUNK, B_SZ), blk, 0, stream>>>(
        delta, u, xdbl, A_log, chP, chH);
    scan_combine_kernel<<<(B_SZ * DINNER) / 256, blk, 0, stream>>>(chP, chH, hst);
    scan_output_kernel<<<dim3(DINNER / 256, NCHUNK, B_SZ), blk, 0, stream>>>(
        delta, u, xdbl, A_log, hst, xz, Dvec, yg_bf);

    // 8) out_proj: out[4096,1024] = yg * out_proj_w^T
    gemm_nt_bf16<0><<<dim3(DMODEL / BN, NTOK / BM), blk, 0, stream>>>(
        yg_bf, DINNER, wo_bf, DINNER, out, DMODEL, nullptr, NTOK, DMODEL, DINNER);
}
